// HODA_20469814133192
// MI455X (gfx1250) — matrix operand order repaired and checked by reading, not compiled, not run on hardware
//
#include <hip/hip_runtime.h>
#include <hip/hip_bf16.h>

typedef __attribute__((ext_vector_type(16))) _Float16 v16h;
typedef __attribute__((ext_vector_type(8)))  _Float16 v8h;
typedef __attribute__((ext_vector_type(8)))  float    v8f;

#define B_ 8
#define V_ 32
#define D_ 256

// ---------------------------------------------------------------- elementwise
__global__ void f32_to_f16_kernel(const float* __restrict__ in,
                                  _Float16* __restrict__ out, int n) {
  int i = blockIdx.x * 256 + threadIdx.x;
  if (i < n) out[i] = (_Float16)in[i];
}

__global__ void convw_kernel(const float* __restrict__ q, const float* __restrict__ k,
                             const float* __restrict__ v, const float* __restrict__ o,
                             _Float16* __restrict__ qh, _Float16* __restrict__ kh,
                             _Float16* __restrict__ vh, _Float16* __restrict__ oh) {
  int i = blockIdx.x * 256 + threadIdx.x;  // 65536 elements exactly
  qh[i] = (_Float16)q[i];
  kh[i] = (_Float16)k[i];
  vh[i] = (_Float16)v[i];
  oh[i] = (_Float16)o[i];
}

// AvgPool2d((3,1),(2,1),(1,0)) along L, count_include_pad (/3 always)
__global__ void pool_kernel(const float* __restrict__ in, float* __restrict__ outf,
                            _Float16* __restrict__ outh, int Lin, int Lout) {
  int idx = blockIdx.x * 256 + threadIdx.x;
  int n = B_ * Lout * V_ * D_;
  if (idx >= n) return;
  int d  = idx & 255;
  int v  = (idx >> 8) & 31;
  int lo = (idx >> 13) % Lout;
  int b  = idx / (Lout * V_ * D_);
  float s = 0.0f;
#pragma unroll
  for (int t = -1; t <= 1; ++t) {
    int li = 2 * lo + t;
    if (li >= 0 && li < Lin)
      s += in[(((size_t)b * Lin + li) * V_ + v) * D_ + d];
  }
  float r = s * (1.0f / 3.0f);
  outf[idx] = r;
  outh[idx] = (_Float16)r;
}

// bilinear interp along L (align_corners=False) back to 512 and accumulate
__global__ void accum_interp_kernel(const float* __restrict__ Y,
                                    float* __restrict__ out, int Lin) {
  int idx = blockIdx.x * 256 + threadIdx.x;  // 2^25 elements exactly
  int d = idx & 255;
  int v = (idx >> 8) & 31;
  int l = (idx >> 13) & 511;
  int b = idx >> 22;
  float scale = (float)Lin * (1.0f / 512.0f);
  float src = (l + 0.5f) * scale - 0.5f;
  src = fminf(fmaxf(src, 0.0f), (float)(Lin - 1));
  int i0 = (int)src;                 // src >= 0 -> trunc == floor
  int i1 = min(i0 + 1, Lin - 1);
  float w = src - (float)i0;
  float y0 = Y[(((size_t)(b * Lin + i0)) * V_ + v) * D_ + d];
  float y1 = Y[(((size_t)(b * Lin + i1)) * V_ + v) * D_ + d];
  out[idx] += y0 * (1.0f - w) + y1 * w;
}

// ---------------------------------------------------------------- WMMA GEMMs
// Y[r,o] = sum_d X[r,d] * W[o,d] + bias[o]
// block = 4 waves, each wave: 16 rows x 64 cols; K=256 in steps of 32.
// mode 0: out[((b*32+v)*L + l)*256 + o]  (Q/K: [B,V,L,D])
// mode 1: out[((b*32+v)*256 + o)*L + l]  (V transposed: [B,V,D,L])
__global__ void __launch_bounds__(128)
gemm_qkv_kernel(const _Float16* __restrict__ X, const _Float16* __restrict__ W,
                const float* __restrict__ bias, _Float16* __restrict__ out,
                int L, int mode) {
  int lane  = threadIdx.x & 31;
  int nbase = (threadIdx.x >> 5) * 64;
  int mtile = blockIdx.x;
  int mrow  = lane & 15;
  int khalf = lane >> 4;
  const _Float16* xr = X + ((size_t)(mtile * 16 + mrow)) * D_;
  v8f acc[4] = {};
  for (int kb = 0; kb < D_; kb += 32) {
    v8h lo = *(const v8h*)(xr + kb + khalf * 16);
    v8h hi = *(const v8h*)(xr + kb + khalf * 16 + 8);
    v16h a;
#pragma unroll
    for (int i = 0; i < 8; i++) { a[i] = lo[i]; a[8 + i] = hi[i]; }
#pragma unroll
    for (int nt = 0; nt < 4; nt++) {
      const _Float16* wr = W + ((size_t)(nbase + nt * 16 + mrow)) * D_ + kb + khalf * 16;
      v16h bf = *(const v16h*)wr;
      acc[nt] = __builtin_amdgcn_wmma_f32_16x16x32_f16(false, a, false, bf,
                                                       (short)0, acc[nt], false, false);
    }
  }
#pragma unroll
  for (int nt = 0; nt < 4; nt++) {
#pragma unroll
    for (int p = 0; p < 8; p++) {
      int row = mtile * 16 + p + khalf * 8;
      int o   = nbase + nt * 16 + (lane & 15);
      float val = acc[nt][p] + bias[o];
      int b = (row >> 5) / L;
      int l = (row >> 5) % L;
      int v = row & 31;
      size_t oi = (mode == 0)
          ? (((size_t)(b * V_ + v)) * L + l) * D_ + o
          : (((size_t)(b * V_ + v)) * D_ + o) * (size_t)L + l;
      out[oi] = (_Float16)val;
    }
  }
}

// Output projection, f32 result written row-major [M,256]
__global__ void __launch_bounds__(128)
gemm_out_kernel(const _Float16* __restrict__ X, const _Float16* __restrict__ W,
                const float* __restrict__ bias, float* __restrict__ out) {
  int lane  = threadIdx.x & 31;
  int nbase = (threadIdx.x >> 5) * 64;
  int mtile = blockIdx.x;
  int mrow  = lane & 15;
  int khalf = lane >> 4;
  const _Float16* xr = X + ((size_t)(mtile * 16 + mrow)) * D_;
  v8f acc[4] = {};
  for (int kb = 0; kb < D_; kb += 32) {
    v8h lo = *(const v8h*)(xr + kb + khalf * 16);
    v8h hi = *(const v8h*)(xr + kb + khalf * 16 + 8);
    v16h a;
#pragma unroll
    for (int i = 0; i < 8; i++) { a[i] = lo[i]; a[8 + i] = hi[i]; }
#pragma unroll
    for (int nt = 0; nt < 4; nt++) {
      const _Float16* wr = W + ((size_t)(nbase + nt * 16 + mrow)) * D_ + kb + khalf * 16;
      v16h bf = *(const v16h*)wr;
      acc[nt] = __builtin_amdgcn_wmma_f32_16x16x32_f16(false, a, false, bf,
                                                       (short)0, acc[nt], false, false);
    }
  }
#pragma unroll
  for (int nt = 0; nt < 4; nt++) {
#pragma unroll
    for (int p = 0; p < 8; p++) {
      size_t row = (size_t)mtile * 16 + p + khalf * 8;
      int o = nbase + nt * 16 + (lane & 15);
      out[row * D_ + o] = acc[nt][p] + bias[o];
    }
  }
}

// ---------------------------------------------------------------- attention
// One wave per (b, v, 16-query-row tile).  Flash-style over key tiles of 32.
// Q: [B,V,L,256] f16, K: [B,V,L,256] f16, Vt: [B,V,256,L] f16 (transposed).
// Oh out: rows [B,L,V,256] f16 to feed the output-projection GEMM.
__global__ void __launch_bounds__(32)
attn_kernel(const _Float16* __restrict__ Qh, const _Float16* __restrict__ Kh,
            const _Float16* __restrict__ Vt, _Float16* __restrict__ Oh, int L) {
  __shared__ float sP[16][33];  // padded: P tile staging for f32->f16 A-layout repack
  int lane  = threadIdx.x;
  int nq    = L >> 4;
  int qt    = blockIdx.x % nq;
  int bv    = blockIdx.x / nq;
  int b     = bv >> 5;
  int v     = bv & 31;
  const _Float16* Qb = Qh + (size_t)bv * L * D_;
  const _Float16* Kb = Kh + (size_t)bv * L * D_;
  const _Float16* Vb = Vt + (size_t)bv * D_ * L;
  int qbase = qt * 16;
  int mrow  = lane & 15;
  int khalf = lane >> 4;
  const float scale = 0.0625f;  // 1/sqrt(256)

  v8f  oacc[16] = {};
  float mrun[8], lrun[8];
#pragma unroll
  for (int p = 0; p < 8; p++) { mrun[p] = -__builtin_inff(); lrun[p] = 0.0f; }

  for (int kb = 0; kb < L; kb += 32) {
    v8f S0 = {}, S1 = {};
#pragma unroll
    for (int dk = 0; dk < 8; dk++) {
      int db = dk * 32;
      const _Float16* qr = Qb + ((size_t)(qbase + mrow)) * D_ + db;
      v8h lo = *(const v8h*)(qr + khalf * 16);
      v8h hi = *(const v8h*)(qr + khalf * 16 + 8);
      v16h a;
#pragma unroll
      for (int i = 0; i < 8; i++) { a[i] = lo[i]; a[8 + i] = hi[i]; }
      v16h b0 = *(const v16h*)(Kb + ((size_t)(kb + mrow)) * D_ + db + khalf * 16);
      S0 = __builtin_amdgcn_wmma_f32_16x16x32_f16(false, a, false, b0, (short)0, S0, false, false);
      v16h b1 = *(const v16h*)(Kb + ((size_t)(kb + 16 + mrow)) * D_ + db + khalf * 16);
      S1 = __builtin_amdgcn_wmma_f32_16x16x32_f16(false, a, false, b1, (short)0, S1, false, false);
    }
    // off-diagonal mask + online softmax (row reductions across the 16-lane half)
#pragma unroll
    for (int p = 0; p < 8; p++) {
      int M = p + khalf * 8;
      int q = qbase + M;
      float s0 = S0[p] * scale, s1 = S1[p] * scale;
      int k0 = kb + (lane & 15), k1 = k0 + 16;
      if (q == k0) s0 = -__builtin_inff();
      if (q == k1) s1 = -__builtin_inff();
      float ml = fmaxf(s0, s1);
#pragma unroll
      for (int off = 1; off < 16; off <<= 1) ml = fmaxf(ml, __shfl_xor(ml, off, 32));
      float mnew  = fmaxf(mrun[p], ml);
      float alpha = __expf(mrun[p] - mnew);
      float p0 = __expf(s0 - mnew);
      float p1 = __expf(s1 - mnew);
      float rs = p0 + p1;
#pragma unroll
      for (int off = 1; off < 16; off <<= 1) rs += __shfl_xor(rs, off, 32);
      lrun[p] = lrun[p] * alpha + rs;
      mrun[p] = mnew;
#pragma unroll
      for (int t = 0; t < 16; t++) oacc[t][p] *= alpha;
      sP[M][lane & 15]        = p0;
      sP[M][16 + (lane & 15)] = p1;
    }
    __syncthreads();
    // repack P tile (C layout in LDS) into f16 A-fragment layout
    v16h pf;
#pragma unroll
    for (int i = 0; i < 8; i++) {
      pf[i]     = (_Float16)sP[mrow][khalf * 16 + i];
      pf[8 + i] = (_Float16)sP[mrow][khalf * 16 + 8 + i];
    }
    // O += P(16x32) * V(32x16 per d-tile); Vt transposed -> contiguous B loads
#pragma unroll
    for (int t = 0; t < 16; t++) {
      v16h vf = *(const v16h*)(Vb + ((size_t)(t * 16 + (lane & 15))) * L + kb + khalf * 16);
      oacc[t] = __builtin_amdgcn_wmma_f32_16x16x32_f16(false, pf, false, vf,
                                                       (short)0, oacc[t], false, false);
    }
    __syncthreads();
  }
  // normalize and store as rows [B,L,V,D]
#pragma unroll
  for (int p = 0; p < 8; p++) {
    float inv = 1.0f / lrun[p];
    int q = qbase + p + khalf * 8;
    size_t base = (((size_t)(b * L + q)) * V_ + v) * D_;
#pragma unroll
    for (int t = 0; t < 16; t++)
      Oh[base + t * 16 + (lane & 15)] = (_Float16)(oacc[t][p] * inv);
  }
}

// ---------------------------------------------------------------- host
extern "C" void kernel_launch(void* const* d_in, const int* in_sizes, int n_in,
                              void* d_out, int out_size, void* d_ws, size_t ws_size,
                              hipStream_t stream) {
  const float* x  = (const float*)d_in[0];
  const float* Wq = (const float*)d_in[1];
  const float* bq = (const float*)d_in[2];
  const float* Wk = (const float*)d_in[3];
  const float* bk = (const float*)d_in[4];
  const float* Wv = (const float*)d_in[5];
  const float* bv = (const float*)d_in[6];
  const float* Wo = (const float*)d_in[7];
  const float* bo = (const float*)d_in[8];
  float* out = (float*)d_out;

  char* p = (char*)d_ws;
  auto take = [&](size_t bytes) { char* r = p; p += (bytes + 255) & ~(size_t)255; return r; };
  float*    x1f = (float*)take((size_t)B_ * 256 * V_ * D_ * 4);
  float*    x2f = (float*)take((size_t)B_ * 128 * V_ * D_ * 4);
  _Float16* Xh  = (_Float16*)take((size_t)B_ * 512 * V_ * D_ * 2);
  _Float16* Qh  = (_Float16*)take((size_t)B_ * 512 * V_ * D_ * 2);
  _Float16* Kh  = (_Float16*)take((size_t)B_ * 512 * V_ * D_ * 2);
  _Float16* Vt  = (_Float16*)take((size_t)B_ * 512 * V_ * D_ * 2);
  _Float16* Oh  = (_Float16*)take((size_t)B_ * 512 * V_ * D_ * 2);
  float*    Yl  = (float*)take((size_t)B_ * 256 * V_ * D_ * 4);
  _Float16* Whq = (_Float16*)take(65536 * 2);
  _Float16* Whk = (_Float16*)take(65536 * 2);
  _Float16* Whv = (_Float16*)take(65536 * 2);
  _Float16* Who = (_Float16*)take(65536 * 2);

  for (int layer = 0; layer < 3; ++layer) {
    int L = 512 >> layer;
    int M = B_ * L * V_;
    int n = M * D_;
    if (layer == 0) {
      f32_to_f16_kernel<<<n / 256, 256, 0, stream>>>(x, Xh, n);
    } else if (layer == 1) {
      pool_kernel<<<n / 256, 256, 0, stream>>>(x, x1f, Xh, 512, 256);
    } else {
      pool_kernel<<<n / 256, 256, 0, stream>>>(x1f, x2f, Xh, 256, 128);
    }
    convw_kernel<<<65536 / 256, 256, 0, stream>>>(
        Wq + (size_t)layer * 65536, Wk + (size_t)layer * 65536,
        Wv + (size_t)layer * 65536, Wo + (size_t)layer * 65536,
        Whq, Whk, Whv, Who);
    gemm_qkv_kernel<<<M / 16, 128, 0, stream>>>(Xh, Whq, bq + layer * 256, Qh, L, 0);
    gemm_qkv_kernel<<<M / 16, 128, 0, stream>>>(Xh, Whk, bk + layer * 256, Kh, L, 0);
    gemm_qkv_kernel<<<M / 16, 128, 0, stream>>>(Xh, Whv, bv + layer * 256, Vt, L, 1);
    attn_kernel<<<B_ * V_ * (L / 16), 32, 0, stream>>>(Qh, Kh, Vt, Oh, L);
    float* ydst = (layer == 0) ? out : Yl;
    gemm_out_kernel<<<M / 16, 128, 0, stream>>>(Oh, Who, bo + layer * 256, ydst);
    if (layer > 0) {
      accum_interp_kernel<<<(1 << 25) / 256, 256, 0, stream>>>(Yl, out, L);
    }
  }
}
